// LanguageModel_49314814492958
// MI455X (gfx1250) — matrix operand order repaired and checked by reading, not compiled, not run on hardware
//
#include <hip/hip_runtime.h>

typedef __bf16 bf16;
typedef __attribute__((ext_vector_type(16))) __bf16 v16bf;
typedef __attribute__((ext_vector_type(8)))  __bf16 v8bf;
typedef __attribute__((ext_vector_type(8)))  float  v8f;

#define LAYERS 8
#define CDIM   1024
#define HEADS  16
#define HDIM   64
#define SEQ    1024
#define BATCHN 4
#define MROWS  (BATCHN*SEQ)
#define VOCAB  32000
#define FFDIM  (4*CDIM)

// B-panel staged in LDS: 64 rows x 64 bytes, padded to 80B stride (20-bank stride
// -> the 16 lanes of a fragment read hit 16 distinct banks). 80 % 16 == 0 keeps
// every b128 access 16B-aligned.
#define BPAN_STRIDE 80

// ---------- helpers ----------

__device__ __forceinline__ bf16 f2bf(float f) {
  unsigned int u = __builtin_bit_cast(unsigned int, f);
  unsigned int r = u + 0x7FFFu + ((u >> 16) & 1u);   // round-to-nearest-even
  unsigned short h = (unsigned short)(r >> 16);
  return __builtin_bit_cast(bf16, h);
}

union ABFrag { v16bf v; v8bf h[2]; };

// A-matrix 16x32 bf16 fragment (this code's contiguous k order, not the ISA table order; B loads match):
// lane m = lane&15; lanes 0-15 hold K=0..15, lanes 16-31 hold K=16..31 (contiguous per lane)
__device__ __forceinline__ v16bf load_a_frag(const bf16* base, int lda, int lane) {
  int m = lane & 15, half = lane >> 4;
  const bf16* p = base + (size_t)m * lda + half * 16;
  ABFrag u;
  u.h[0] = *(const v8bf*)(p);
  u.h[1] = *(const v8bf*)(p + 8);
  return u.v;
}

// B-matrix 32x16 bf16 fragment from N-major ("Bt") storage: Bt[n][k], ldb = K.
// lane n = lane&15; lanes 0-15 hold K=0..15, lanes 16-31 hold K=16..31 (contiguous per lane)
__device__ __forceinline__ v16bf load_b_frag(const bf16* base, int ldb, int lane) {
  int n = lane & 15, half = lane >> 4;
  const bf16* p = base + (size_t)n * ldb + half * 16;
  return *(const v16bf*)(p);
}

// B fragment from an LDS-staged panel (row = 16j+n, 32B per lane, 80B row stride)
__device__ __forceinline__ v16bf load_b_lds(const char* panel, int j, int lane) {
  int n = lane & 15, half = lane >> 4;
  const bf16* p = (const bf16*)(panel + (16 * j + n) * BPAN_STRIDE + half * 32);
  ABFrag u;
  u.h[0] = *(const v8bf*)(p);
  u.h[1] = *(const v8bf*)(p + 8);
  return u.v;
}

__device__ __forceinline__ v8f wmma_bf16(v16bf a, v16bf b, v8f c) {
  return __builtin_amdgcn_wmma_f32_16x16x32_bf16(false, a, false, b, (short)0, c, false, false);
}

// ---------- weight transpose/convert: fp32 [batch][K][N] -> bf16 [batch][N][K] ----------

__global__ __launch_bounds__(256) void transpose_to_bf16(
    const float* __restrict__ in, bf16* __restrict__ out,
    int K, int N, long long total)
{
  long long i = (long long)blockIdx.x * blockDim.x + threadIdx.x;
  long long stride = (long long)gridDim.x * blockDim.x;
  long long KN = (long long)K * N;
  for (; i < total; i += stride) {
    long long batch = i / KN;
    long long rem = i - batch * KN;
    int n = (int)(rem / K);
    int kk = (int)(rem - (long long)n * K);
    out[i] = f2bf(in[batch * KN + (long long)kk * N + n]);
  }
}

// ---------- embedding: x[m][c] = tok[idx[m]][c] + pos[m%T][c] ----------

__global__ __launch_bounds__(256) void embed_kernel(
    const int* __restrict__ idx, const float* __restrict__ tok,
    const float* __restrict__ pos, float* __restrict__ x)
{
  int m = blockIdx.x;
  int t = m & (SEQ - 1);
  int id = idx[m];
  for (int c = threadIdx.x; c < CDIM; c += 256)
    x[(size_t)m * CDIM + c] = tok[(size_t)id * CDIM + c] + pos[(size_t)t * CDIM + c];
}

// ---------- layernorm: fp32 in -> bf16 out (one block per row, C=1024) ----------

__global__ __launch_bounds__(256) void layernorm_to_bf16(
    const float* __restrict__ x, const float* __restrict__ g,
    const float* __restrict__ bta, bf16* __restrict__ out)
{
  __shared__ float red[8];
  int row = blockIdx.x;
  const float* xr = x + (size_t)row * CDIM;
  float lx[4];
  float s = 0.f;
#pragma unroll
  for (int i = 0; i < 4; ++i) { lx[i] = xr[threadIdx.x + i * 256]; s += lx[i]; }
#pragma unroll
  for (int off = 16; off; off >>= 1) s += __shfl_xor(s, off, 32);
  if ((threadIdx.x & 31) == 0) red[threadIdx.x >> 5] = s;
  __syncthreads();
  float mean = 0.f;
#pragma unroll
  for (int w = 0; w < 8; ++w) mean += red[w];
  mean *= (1.0f / CDIM);
  __syncthreads();
  float v = 0.f;
#pragma unroll
  for (int i = 0; i < 4; ++i) { float d = lx[i] - mean; v += d * d; }
#pragma unroll
  for (int off = 16; off; off >>= 1) v += __shfl_xor(v, off, 32);
  if ((threadIdx.x & 31) == 0) red[threadIdx.x >> 5] = v;
  __syncthreads();
  float var = 0.f;
#pragma unroll
  for (int w = 0; w < 8; ++w) var += red[w];
  var *= (1.0f / CDIM);
  float rstd = rsqrtf(var + 1e-5f);
#pragma unroll
  for (int i = 0; i < 4; ++i) {
    int c = threadIdx.x + i * 256;
    out[(size_t)row * CDIM + c] = f2bf((lx[i] - mean) * rstd * g[c] + bta[c]);
  }
}

// ---------- generic WMMA GEMM: D = A[M,K](bf16) * Bt[N,K](bf16) ----------
// B panels are double-buffered in LDS via GLOBAL_LOAD_ASYNC_TO_LDS_B128 (ASYNCcnt):
// one 4KB panel per K-step == 256 threads x one b128 async copy.
// MODE 0: bf16 scatter -> [B,H,T,HD]   (q, k)
// MODE 1: bf16 scatter -> [B,H,HD,T]   (vT)
// MODE 2: fp32 out[m][n] = resid + acc + bias[n]
// MODE 3: bf16 out[m][n] = gelu(acc + bias[n])
// MODE 4: fp32 out[m][n] = acc + bias[n]
template<int MODE>
__global__ __launch_bounds__(256) void gemm_wmma(
    const bf16* __restrict__ A, const bf16* __restrict__ Bt,
    const float* __restrict__ bias, const float* __restrict__ resid,
    void* __restrict__ outp, int N, int K)
{
  __shared__ __align__(16) char bpan[2][64 * BPAN_STRIDE];

  const int tid  = threadIdx.x;
  const int lane = tid & 31;
  const int wave = tid >> 5;
  const int row0 = blockIdx.y * 128 + wave * 16;
  const int col0 = blockIdx.x * 64;

  // async-copy assignment: thread -> (row, 16B segment) of the 64x64B panel
  const int srow = tid >> 2;
  const int sseg = tid & 3;
  const bf16* gsrc0 = Bt + (size_t)(col0 + srow) * K + sseg * 8;

  v8f acc[4] = {};
  const bf16* arow = A + (size_t)row0 * K;

  // prologue: stage panel for k0 = 0
  {
    unsigned l = (unsigned)(size_t)(&bpan[0][srow * BPAN_STRIDE + sseg * 16]);
    asm volatile("global_load_async_to_lds_b128 %0, %1, off"
                 :: "v"(l), "v"(gsrc0) : "memory");
  }

  int cur = 0;
  for (int k0 = 0; k0 < K; k0 += 32, cur ^= 1) {
    const bool has_next = (k0 + 32) < K;     // uniform across block
    if (has_next) {
      unsigned l = (unsigned)(size_t)(&bpan[cur ^ 1][srow * BPAN_STRIDE + sseg * 16]);
      const bf16* g = gsrc0 + (k0 + 32);
      asm volatile("global_load_async_to_lds_b128 %0, %1, off"
                   :: "v"(l), "v"(g) : "memory");
      asm volatile("s_wait_asynccnt 0x1" ::: "memory");  // panel[cur] done, next in flight
    } else {
      asm volatile("s_wait_asynccnt 0x0" ::: "memory");
    }
    __syncthreads();                         // publish panel[cur] to all waves

    __builtin_prefetch(arow + k0 + 128, 0, 0);
    v16bf a = load_a_frag(arow + k0, K, lane);
#pragma unroll
    for (int j = 0; j < 4; ++j) {
      v16bf b = load_b_lds(&bpan[cur][0], j, lane);
      acc[j] = wmma_bf16(a, b, acc[j]);
    }
    __syncthreads();                         // all reads of panel[cur] done before re-stage
  }

  const int dn = lane & 15;
  const int dm = (lane >> 4) * 8;

  if constexpr (MODE == 0) {
    bf16* dst = (bf16*)outp;
#pragma unroll
    for (int j = 0; j < 4; ++j) {
      int n = col0 + 16 * j + dn;
      int hh = n >> 6, d = n & (HDIM - 1);
#pragma unroll
      for (int r = 0; r < 8; ++r) {
        int m = row0 + dm + r;
        int bb = m >> 10, t = m & (SEQ - 1);
        dst[(((size_t)(bb * HEADS + hh)) * SEQ + t) * HDIM + d] = f2bf(acc[j][r]);
      }
    }
  } else if constexpr (MODE == 1) {
    bf16* dst = (bf16*)outp;
#pragma unroll
    for (int j = 0; j < 4; ++j) {
      int n = col0 + 16 * j + dn;
      int hh = n >> 6, d = n & (HDIM - 1);
#pragma unroll
      for (int r = 0; r < 8; ++r) {
        int m = row0 + dm + r;
        int bb = m >> 10, t = m & (SEQ - 1);
        dst[(((size_t)(bb * HEADS + hh)) * HDIM + d) * SEQ + t] = f2bf(acc[j][r]);
      }
    }
  } else if constexpr (MODE == 2) {
    float* dst = (float*)outp;
#pragma unroll
    for (int j = 0; j < 4; ++j) {
      int n = col0 + 16 * j + dn;
      float bv = bias[n];
#pragma unroll
      for (int r = 0; r < 8; ++r) {
        size_t o = (size_t)(row0 + dm + r) * N + n;
        dst[o] = resid[o] + acc[j][r] + bv;
      }
    }
  } else if constexpr (MODE == 3) {
    bf16* dst = (bf16*)outp;
#pragma unroll
    for (int j = 0; j < 4; ++j) {
      int n = col0 + 16 * j + dn;
      float bv = bias[n];
#pragma unroll
      for (int r = 0; r < 8; ++r) {
        float v = acc[j][r] + bv;
        v = 0.5f * v * (1.0f + erff(v * 0.70710678118f));   // exact GELU
        dst[(size_t)(row0 + dm + r) * N + n] = f2bf(v);
      }
    }
  } else {
    float* dst = (float*)outp;
#pragma unroll
    for (int j = 0; j < 4; ++j) {
      int n = col0 + 16 * j + dn;
      float bv = bias[n];
#pragma unroll
      for (int r = 0; r < 8; ++r)
        dst[(size_t)(row0 + dm + r) * N + n] = acc[j][r] + bv;
    }
  }
}

// ---------- flash attention: one wave per 16-query tile ----------
// q,k: [B,H,T,HD] bf16 row-major; vT: [B,H,HD,T] bf16; z: [B*T, C] bf16

__global__ __launch_bounds__(128) void attn_wmma(
    const bf16* __restrict__ q, const bf16* __restrict__ k,
    const bf16* __restrict__ vT, bf16* __restrict__ z)
{
  __shared__ __align__(32) bf16 lds_p[4][16 * 32];
  const int lane = threadIdx.x & 31;
  const int wave = threadIdx.x >> 5;
  const int gid = blockIdx.x * 4 + wave;
  const int bh = gid >> 6;              // T/16 = 64 tiles per (b,h)
  const int t0 = (gid & 63) << 4;
  const int b = bh >> 4;                // H = 16
  const int h = bh & (HEADS - 1);

  const bf16* qh = q  + (size_t)bh * SEQ * HDIM;
  const bf16* kh = k  + (size_t)bh * SEQ * HDIM;
  const bf16* vh = vT + (size_t)bh * HDIM * SEQ;

  // Q fragments for K-dim = 64 (two k-steps), loaded once
  v16bf a0 = load_a_frag(qh + (size_t)t0 * HDIM,      HDIM, lane);
  v16bf a1 = load_a_frag(qh + (size_t)t0 * HDIM + 32, HDIM, lane);

  v8f acc[4] = {};
  float mrow[8], lrow[8];
#pragma unroll
  for (int r = 0; r < 8; ++r) { mrow[r] = -3.0e38f; lrow[r] = 0.f; }

  const int dn = lane & 15;
  const int dm = (lane >> 4) * 8;
  const float scale = 0.125f;           // HD^-0.5

  for (int s0 = 0; s0 < t0 + 16; s0 += 32) {
    v8f S0 = {}, S1 = {};
    S0 = wmma_bf16(a0, load_b_frag(kh + (size_t)s0 * HDIM,             HDIM, lane), S0);
    S0 = wmma_bf16(a1, load_b_frag(kh + (size_t)s0 * HDIM + 32,        HDIM, lane), S0);
    S1 = wmma_bf16(a0, load_b_frag(kh + (size_t)(s0 + 16) * HDIM,      HDIM, lane), S1);
    S1 = wmma_bf16(a1, load_b_frag(kh + (size_t)(s0 + 16) * HDIM + 32, HDIM, lane), S1);

#pragma unroll
    for (int r = 0; r < 8; ++r) {
      int t = t0 + dm + r;
      float x0 = S0[r] * scale, x1 = S1[r] * scale;
      if (s0 + dn > t)      x0 = -3.0e38f;   // causal mask
      if (s0 + 16 + dn > t) x1 = -3.0e38f;
      // row max across 32 keys (16-lane shuffle reduce matches C/D row layout)
      float mx = fmaxf(x0, x1);
#pragma unroll
      for (int off = 8; off; off >>= 1) mx = fmaxf(mx, __shfl_xor(mx, off, 32));
      float newm = fmaxf(mrow[r], mx);
      float resc = __expf(mrow[r] - newm);
      float p0 = __expf(x0 - newm);
      float p1 = __expf(x1 - newm);
      float rs = p0 + p1;
#pragma unroll
      for (int off = 8; off; off >>= 1) rs += __shfl_xor(rs, off, 32);
      lrow[r] = lrow[r] * resc + rs;
      mrow[r] = newm;
#pragma unroll
      for (int j = 0; j < 4; ++j) acc[j][r] *= resc;
      lds_p[wave][(dm + r) * 32 + dn]      = f2bf(p0);
      lds_p[wave][(dm + r) * 32 + 16 + dn] = f2bf(p1);
    }
    asm volatile("s_wait_dscnt 0" ::: "memory");   // re-swizzle P through LDS into A layout
    v16bf pa = load_a_frag(&lds_p[wave][0], 32, lane);
#pragma unroll
    for (int j = 0; j < 4; ++j) {
      v16bf bv = load_b_frag(vh + (size_t)(16 * j) * SEQ + s0, SEQ, lane);
      acc[j] = wmma_bf16(pa, bv, acc[j]);
    }
  }

#pragma unroll
  for (int j = 0; j < 4; ++j) {
#pragma unroll
    for (int r = 0; r < 8; ++r) {
      int t = t0 + dm + r;
      float v = acc[j][r] / lrow[r];
      z[((size_t)(b * SEQ + t)) * CDIM + h * HDIM + 16 * j + dn] = f2bf(v);
    }
  }
}

// ---------- host orchestration ----------

extern "C" void kernel_launch(void* const* d_in, const int* in_sizes, int n_in,
                              void* d_out, int out_size, void* d_ws, size_t ws_size,
                              hipStream_t stream)
{
  (void)in_sizes; (void)n_in; (void)out_size; (void)ws_size;

  const int*   idx   = (const int*)d_in[0];
  const float* tok   = (const float*)d_in[1];
  const float* pos   = (const float*)d_in[2];
  const float* Wq    = (const float*)d_in[3];
  const float* Wk    = (const float*)d_in[4];
  const float* Wv    = (const float*)d_in[5];
  const float* Wo    = (const float*)d_in[6];
  const float* bo    = (const float*)d_in[7];
  const float* ln1g  = (const float*)d_in[8];
  const float* ln1b  = (const float*)d_in[9];
  const float* ln2g  = (const float*)d_in[10];
  const float* ln2b  = (const float*)d_in[11];
  const float* W1    = (const float*)d_in[12];
  const float* b1    = (const float*)d_in[13];
  const float* W2    = (const float*)d_in[14];
  const float* b2    = (const float*)d_in[15];
  const float* lnfg  = (const float*)d_in[16];
  const float* lnfb  = (const float*)d_in[17];
  const float* Wout  = (const float*)d_in[18];
  const float* bout  = (const float*)d_in[19];

  char* ws = (char*)d_ws;
  size_t off = 0;
  auto alloc = [&](size_t bytes) -> char* {
    char* p = ws + off;
    off = (off + bytes + 255) & ~(size_t)255;
    return p;
  };

  bf16* qBt    = (bf16*)alloc((size_t)LAYERS * CDIM * CDIM * 2);
  bf16* kBt    = (bf16*)alloc((size_t)LAYERS * CDIM * CDIM * 2);
  bf16* vBt    = (bf16*)alloc((size_t)LAYERS * CDIM * CDIM * 2);
  bf16* WoBt   = (bf16*)alloc((size_t)LAYERS * CDIM * CDIM * 2);
  bf16* W1Bt   = (bf16*)alloc((size_t)LAYERS * CDIM * FFDIM * 2);
  bf16* W2Bt   = (bf16*)alloc((size_t)LAYERS * FFDIM * CDIM * 2);
  bf16* WoutBt = (bf16*)alloc((size_t)CDIM * VOCAB * 2);
  float* xbuf  = (float*)alloc((size_t)MROWS * CDIM * 4);
  bf16* hbuf   = (bf16*)alloc((size_t)MROWS * CDIM * 2);
  bf16* qb     = (bf16*)alloc((size_t)MROWS * CDIM * 2);
  bf16* kb     = (bf16*)alloc((size_t)MROWS * CDIM * 2);
  bf16* vTb    = (bf16*)alloc((size_t)MROWS * CDIM * 2);
  bf16* zb     = (bf16*)alloc((size_t)MROWS * CDIM * 2);
  bf16* ffb    = (bf16*)alloc((size_t)MROWS * FFDIM * 2);

  auto tconv = [&](const float* src, bf16* dst, long long batch, int K, int N) {
    long long total = batch * (long long)K * N;
    long long blocks = (total + 255) / 256;
    int grid = (int)(blocks > 65535 ? 65535 : blocks);
    transpose_to_bf16<<<dim3(grid), dim3(256), 0, stream>>>(src, dst, K, N, total);
  };

  // weight conversion (fp32 [K,N] -> bf16 [N,K], batched)
  tconv(Wq,   qBt,    (long long)LAYERS * HEADS, CDIM,  HDIM);
  tconv(Wk,   kBt,    (long long)LAYERS * HEADS, CDIM,  HDIM);
  tconv(Wv,   vBt,    (long long)LAYERS * HEADS, CDIM,  HDIM);
  tconv(Wo,   WoBt,   LAYERS,                    CDIM,  CDIM);
  tconv(W1,   W1Bt,   LAYERS,                    CDIM,  FFDIM);
  tconv(W2,   W2Bt,   LAYERS,                    FFDIM, CDIM);
  tconv(Wout, WoutBt, 1,                         CDIM,  VOCAB);

  embed_kernel<<<dim3(MROWS), dim3(256), 0, stream>>>(idx, tok, pos, xbuf);

  const dim3 blk(256);
  for (int l = 0; l < LAYERS; ++l) {
    const bf16* qBl  = qBt  + (size_t)l * CDIM * CDIM;
    const bf16* kBl  = kBt  + (size_t)l * CDIM * CDIM;
    const bf16* vBl  = vBt  + (size_t)l * CDIM * CDIM;
    const bf16* oBl  = WoBt + (size_t)l * CDIM * CDIM;
    const bf16* w1Bl = W1Bt + (size_t)l * CDIM * FFDIM;
    const bf16* w2Bl = W2Bt + (size_t)l * FFDIM * CDIM;

    layernorm_to_bf16<<<dim3(MROWS), blk, 0, stream>>>(xbuf, ln1g + l * CDIM, ln1b + l * CDIM, hbuf);

    gemm_wmma<0><<<dim3(CDIM / 64, MROWS / 128), blk, 0, stream>>>(hbuf, qBl, nullptr, nullptr, qb,  CDIM, CDIM);
    gemm_wmma<0><<<dim3(CDIM / 64, MROWS / 128), blk, 0, stream>>>(hbuf, kBl, nullptr, nullptr, kb,  CDIM, CDIM);
    gemm_wmma<1><<<dim3(CDIM / 64, MROWS / 128), blk, 0, stream>>>(hbuf, vBl, nullptr, nullptr, vTb, CDIM, CDIM);

    attn_wmma<<<dim3((BATCHN * HEADS * (SEQ / 16)) / 4), dim3(128), 0, stream>>>(qb, kb, vTb, zb);

    gemm_wmma<2><<<dim3(CDIM / 64, MROWS / 128), blk, 0, stream>>>(zb, oBl, bo + l * CDIM, xbuf, xbuf, CDIM, CDIM);

    layernorm_to_bf16<<<dim3(MROWS), blk, 0, stream>>>(xbuf, ln2g + l * CDIM, ln2b + l * CDIM, hbuf);

    gemm_wmma<3><<<dim3(FFDIM / 64, MROWS / 128), blk, 0, stream>>>(hbuf, w1Bl, b1 + l * FFDIM, nullptr, ffb, FFDIM, CDIM);
    gemm_wmma<2><<<dim3(CDIM / 64, MROWS / 128), blk, 0, stream>>>(ffb, w2Bl, b2 + l * CDIM, xbuf, xbuf, CDIM, FFDIM);
  }

  layernorm_to_bf16<<<dim3(MROWS), blk, 0, stream>>>(xbuf, lnfg, lnfb, hbuf);
  gemm_wmma<4><<<dim3(VOCAB / 64, MROWS / 128), blk, 0, stream>>>(hbuf, WoutBt, bout, nullptr, d_out, VOCAB, CDIM);
}
